// ResidualEquivariantModule_65309272703451
// MI455X (gfx1250) — matrix operand order repaired and checked by reading, not compiled, not run on hardware
//
#include <hip/hip_runtime.h>
#include <hip/hip_bf16.h>

typedef __attribute__((ext_vector_type(16))) _Float16 v16h;
typedef __attribute__((ext_vector_type(8)))  _Float16 v8h;
typedef __attribute__((ext_vector_type(4)))  _Float16 h4;
typedef __attribute__((ext_vector_type(8)))  float    v8f;
typedef __attribute__((ext_vector_type(4)))  float    v4f;

#define NB   4
#define NN   4096
#define NK   32
#define NC   64
#define NPTS (NB * NN)          // 16384 points
#define RADIUS_F 0.4f
#define EPS_F    0.01f

// ---------------------------------------------------------------------------
// Kernel 0: Wcv = Wc @ Wv (f32 in LDS), then emit Wcv and Ws as f16 in
// WMMA B-fragment order: frag index f = ((ki*4+ct)*32 + lane)*16 + h
//   K = ki*32 + (h>>3)*16 + (lane>>4)*8 + (h&7) ;  n = ct*16 + (lane&15)
// so each lane's 16 halves are contiguous (32B) in memory.
// ---------------------------------------------------------------------------
__global__ __launch_bounds__(256) void k_prep_weights(
    const float* __restrict__ Wc, const float* __restrict__ Wv,
    const float* __restrict__ Ws,
    _Float16* __restrict__ cvfrag, _Float16* __restrict__ sfrag)
{
    __shared__ float cv[64 * 64];
    for (int e = threadIdx.x; e < 4096; e += 256) {
        int r = e >> 6, c = e & 63;
        float s = 0.f;
        #pragma unroll 8
        for (int k = 0; k < 64; ++k) s += Wc[r * 64 + k] * Wv[k * 64 + c];
        cv[e] = s;
    }
    __syncthreads();
    for (int f = threadIdx.x; f < 4096; f += 256) {
        int h    = f & 15;
        int lane = (f >> 4) & 31;
        int ct   = (f >> 9) & 3;
        int ki   = (f >> 11) & 1;
        int K = ki * 32 + (h >> 3) * 16 + (lane >> 4) * 8 + (h & 7);
        int n = ct * 16 + (lane & 15);
        cvfrag[f] = (_Float16)cv[K * 64 + n];
        sfrag[f]  = (_Float16)Ws[K * 64 + n];
    }
}

// ---------------------------------------------------------------------------
// Kernel 1: gather-aggregate (the L2-bandwidth-bound phase).
// 16 threads per point, each owning a channel quad (float4 / b128 loads).
// Each thread computes 2 of the 32 neighbor weights into LDS, then all 16
// accumulate the 9 m-components over 32 neighbors with wide loads.
// Emits f16 Y rows and masked f16 X rows (skip-GEMM A operands).
// ---------------------------------------------------------------------------
__global__ __launch_bounds__(256) void k_aggregate(
    const float* __restrict__ x0, const float* __restrict__ x1,
    const float* __restrict__ x2,
    const float* __restrict__ points, const int* __restrict__ pidx,
    const float* __restrict__ mask,
    _Float16* __restrict__ Y0, _Float16* __restrict__ Y1,
    _Float16* __restrict__ Y2,
    _Float16* __restrict__ X0, _Float16* __restrict__ X1,
    _Float16* __restrict__ X2)
{
    __shared__ float wsh[16][NK];
    __shared__ int   ish[16][NK];
    const int g = threadIdx.x >> 4;        // point within block (0..15)
    const int t = threadIdx.x & 15;        // channel quad
    const int p = blockIdx.x * 16 + g;     // flat point id = b*NN + n
    const int b = p >> 12;                 // NN = 4096

    const float cx = points[p * 3 + 0];
    const float cy = points[p * 3 + 1];
    const float cz = points[p * 3 + 2];
    #pragma unroll
    for (int kk = 0; kk < 2; ++kk) {
        int k  = t + kk * 16;
        int id = pidx[p * NK + k];
        int q  = b * NN + id;
        float dx = points[q * 3 + 0] - cx;
        float dy = points[q * 3 + 1] - cy;
        float dz = points[q * 3 + 2] - cz;
        float r = sqrtf(dx * dx + dy * dy + dz * dz + 1e-12f);
        float w = 1.f - r * (1.f / RADIUS_F);
        w = w > 0.f ? w : 0.f;
        w *= mask[q];                      // neighbor mask folded into weight
        wsh[g][k] = w;
        ish[g][k] = id;
    }
    __syncthreads();

    v4f acc[9];
    #pragma unroll
    for (int m = 0; m < 9; ++m)
        #pragma unroll
        for (int i = 0; i < 4; ++i) acc[m][i] = 0.f;

    const int c4 = t * 4;
    for (int k = 0; k < NK; ++k) {
        float w = wsh[g][k];
        int q = b * NN + ish[g][k];
        {
            v4f xv = *(const v4f*)(x0 + q * 64 + c4);
            #pragma unroll
            for (int i = 0; i < 4; ++i) acc[0][i] += w * xv[i];
        }
        #pragma unroll
        for (int m = 0; m < 3; ++m) {
            v4f xv = *(const v4f*)(x1 + (q * 3 + m) * 64 + c4);
            #pragma unroll
            for (int i = 0; i < 4; ++i) acc[1 + m][i] += w * xv[i];
        }
        #pragma unroll
        for (int m = 0; m < 5; ++m) {
            v4f xv = *(const v4f*)(x2 + (q * 5 + m) * 64 + c4);
            #pragma unroll
            for (int i = 0; i < 4; ++i) acc[4 + m][i] += w * xv[i];
        }
    }

    // f16 Y stores (8B each)
    {
        h4 hv;
        #pragma unroll
        for (int i = 0; i < 4; ++i) hv[i] = (_Float16)acc[0][i];
        *(h4*)(Y0 + p * 64 + c4) = hv;
    }
    #pragma unroll
    for (int m = 0; m < 3; ++m) {
        h4 hv;
        #pragma unroll
        for (int i = 0; i < 4; ++i) hv[i] = (_Float16)acc[1 + m][i];
        *(h4*)(Y1 + (p * 3 + m) * 64 + c4) = hv;
    }
    #pragma unroll
    for (int m = 0; m < 5; ++m) {
        h4 hv;
        #pragma unroll
        for (int i = 0; i < 4; ++i) hv[i] = (_Float16)acc[4 + m][i];
        *(h4*)(Y2 + (p * 5 + m) * 64 + c4) = hv;
    }

    // masked center x -> f16 (skip GEMM A operand)
    const float mm = mask[p];
    {
        v4f xv = *(const v4f*)(x0 + p * 64 + c4);
        h4 hv;
        #pragma unroll
        for (int i = 0; i < 4; ++i) hv[i] = (_Float16)(xv[i] * mm);
        *(h4*)(X0 + p * 64 + c4) = hv;
    }
    #pragma unroll
    for (int m = 0; m < 3; ++m) {
        v4f xv = *(const v4f*)(x1 + (p * 3 + m) * 64 + c4);
        h4 hv;
        #pragma unroll
        for (int i = 0; i < 4; ++i) hv[i] = (_Float16)(xv[i] * mm);
        *(h4*)(X1 + (p * 3 + m) * 64 + c4) = hv;
    }
    #pragma unroll
    for (int m = 0; m < 5; ++m) {
        v4f xv = *(const v4f*)(x2 + (p * 5 + m) * 64 + c4);
        h4 hv;
        #pragma unroll
        for (int i = 0; i < 4; ++i) hv[i] = (_Float16)(xv[i] * mm);
        *(h4*)(X2 + (p * 5 + m) * 64 + c4) = hv;
    }
}

// ---------------------------------------------------------------------------
// Kernel 2 (fused GEMM + epilogue).  Per degree M (WAVES waves, ROWS=WAVES*16
// rows = PTS whole points per block; grids divide exactly so EXEC is full):
//   V = Y @ Wcv ; S = X @ Ws  via v_wmma_f32_16x16x32_f16 (16 per wave),
//   D fragments staged to LDS in per-lane-contiguous order, then the block
//   computes the M-norm / activation / skip / mask epilogue from LDS and
//   writes the final output (no f32 round trip through HBM).
// A fragment layout (ISA 16-bit A 16x32): lane<16 rows M=lane, halves cover
// K {0..7,16..23}; lane>=16 same row, K {8..15,24..31}.
// ---------------------------------------------------------------------------
template <int M, int WAVES>
__global__ __launch_bounds__(WAVES * 32) void k_gemm_fused(
    const _Float16* __restrict__ Y, const _Float16* __restrict__ X,
    const _Float16* __restrict__ Wcvf, const _Float16* __restrict__ Wsf,
    const float* __restrict__ mask, float* __restrict__ out)
{
    constexpr int ROWS = WAVES * 16;
    constexpr int PTS  = ROWS / M;
    __shared__ float Vl[WAVES * 4 * 32 * 8];   // [wave][ct][lane][8]
    __shared__ float Sl[WAVES * 4 * 32 * 8];

    const int wave = threadIdx.x >> 5;
    const int lane = threadIdx.x & 31;
    const int row0 = blockIdx.x * ROWS + wave * 16;

    const int rl = lane & 15;
    const int kh = lane >> 4;              // 0 or 1
    const _Float16* yr = Y + (size_t)(row0 + rl) * 64;
    const _Float16* xr = X + (size_t)(row0 + rl) * 64;

    v8f accV[4], accS[4];
    #pragma unroll
    for (int ct = 0; ct < 4; ++ct) {
        #pragma unroll
        for (int i = 0; i < 8; ++i) { accV[ct][i] = 0.f; accS[ct][i] = 0.f; }
    }

    #pragma unroll
    for (int ki = 0; ki < 2; ++ki) {
        int base = ki * 32 + kh * 8;
        v8h ylo = *(const v8h*)(yr + base);
        v8h yhi = *(const v8h*)(yr + base + 16);
        v8h xlo = *(const v8h*)(xr + base);
        v8h xhi = *(const v8h*)(xr + base + 16);
        v16h aY, aX;
        #pragma unroll
        for (int i = 0; i < 8; ++i) {
            aY[i] = ylo[i]; aY[i + 8] = yhi[i];
            aX[i] = xlo[i]; aX[i + 8] = xhi[i];
        }
        #pragma unroll
        for (int ct = 0; ct < 4; ++ct) {
            v16h bcv = *(const v16h*)(Wcvf + (((ki * 4 + ct) * 32 + lane) << 4));
            v16h bs  = *(const v16h*)(Wsf  + (((ki * 4 + ct) * 32 + lane) << 4));
            accV[ct] = __builtin_amdgcn_wmma_f32_16x16x32_f16(
                false, aY, false, bcv, (short)0, accV[ct], false, false);
            accS[ct] = __builtin_amdgcn_wmma_f32_16x16x32_f16(
                false, aX, false, bs, (short)0, accS[ct], false, false);
        }
    }

    // Stage D fragments to LDS, per-lane contiguous (compiler merges to wide ds)
    #pragma unroll
    for (int ct = 0; ct < 4; ++ct) {
        float* dv = &Vl[((wave * 4 + ct) * 32 + lane) * 8];
        float* ds = &Sl[((wave * 4 + ct) * 32 + lane) * 8];
        #pragma unroll
        for (int i = 0; i < 8; ++i) { dv[i] = accV[ct][i]; ds[i] = accS[ct][i]; }
    }
    __syncthreads();

    // Epilogue: per (local point, co); element (row, co) lives at
    // idx = ((tile*4 + co>>4)*32 + (co&15) + 16*((row&15)>=8))*8 + (row&7)
    const int p0 = blockIdx.x * PTS;
    for (int it = threadIdx.x; it < PTS * 64; it += WAVES * 32) {
        const int pl = it >> 6;
        const int co = it & 63;
        const int ct = co >> 4;
        const int cl = co & 15;
        const float mm = mask[p0 + pl];
        float v[M], s[M];
        float ss = 0.f;
        #pragma unroll
        for (int m = 0; m < M; ++m) {
            int row = pl * M + m;
            int w   = row >> 4;
            int r   = row & 15;
            int idx = ((w * 4 + ct) * 32 + cl + ((r & 8) ? 16 : 0)) * 8 + (r & 7);
            v[m] = Vl[idx];
            s[m] = Sl[idx];
            ss += v[m] * v[m];
        }
        const float inv = 1.f / sqrtf(ss + EPS_F);
        float a[M];
        if (M == 1) {
            float vn = v[0] * inv;
            a[0] = vn > 0.f ? vn : 0.f;
        } else {
            float nn = sqrtf(ss * inv * inv + EPS_F);
            float sg = 1.f / (1.f + __expf(-nn));
            #pragma unroll
            for (int m = 0; m < M; ++m) a[m] = v[m] * inv * sg;
        }
        #pragma unroll
        for (int m = 0; m < M; ++m)
            out[((size_t)(p0 + pl) * M + m) * 64 + co] = (a[m] + s[m]) * mm;
    }
}

// ---------------------------------------------------------------------------
extern "C" void kernel_launch(void* const* d_in, const int* in_sizes, int n_in,
                              void* d_out, int out_size, void* d_ws,
                              size_t ws_size, hipStream_t stream)
{
    (void)in_sizes; (void)n_in; (void)out_size; (void)ws_size;

    const float* x0     = (const float*)d_in[0];
    const float* x1     = (const float*)d_in[1];
    const float* x2     = (const float*)d_in[2];
    const float* points = (const float*)d_in[3];
    const int*   pidx   = (const int*)d_in[4];
    const float* mask   = (const float*)d_in[5];
    const float* Wc[3]  = {(const float*)d_in[6],  (const float*)d_in[7],
                           (const float*)d_in[8]};
    const float* Wv[3]  = {(const float*)d_in[9],  (const float*)d_in[10],
                           (const float*)d_in[11]};
    const float* Wsk[3] = {(const float*)d_in[12], (const float*)d_in[13],
                           (const float*)d_in[14]};

    // -------- workspace carve-out (all sizes 256B-aligned) --------
    constexpr size_t P = NPTS;
    constexpr size_t szY0 = P * 64  * 2;   // f16 rows, M=1
    constexpr size_t szY1 = P * 192 * 2;   // M=3
    constexpr size_t szY2 = P * 320 * 2;   // M=5
    constexpr size_t szWf = 4096 * 2;      // one fragment set (f16)

    char* ws = (char*)d_ws;
    size_t o = 0;
    _Float16* Y0 = (_Float16*)(ws + o); o += szY0;
    _Float16* Y1 = (_Float16*)(ws + o); o += szY1;
    _Float16* Y2 = (_Float16*)(ws + o); o += szY2;
    _Float16* X0 = (_Float16*)(ws + o); o += szY0;
    _Float16* X1 = (_Float16*)(ws + o); o += szY1;
    _Float16* X2 = (_Float16*)(ws + o); o += szY2;
    _Float16* cvf[3]; _Float16* sf[3];
    for (int d = 0; d < 3; ++d) {
        cvf[d] = (_Float16*)(ws + o); o += szWf;
        sf[d]  = (_Float16*)(ws + o); o += szWf;
    }

    float* out0 = (float*)d_out;
    float* out1 = out0 + (size_t)NPTS * 64;        // after (B,N,1,64)
    float* out2 = out0 + (size_t)NPTS * 64 * 4;    // after (B,N,1+3,64)

    // -------- launches --------
    for (int d = 0; d < 3; ++d)
        k_prep_weights<<<1, 256, 0, stream>>>(Wc[d], Wv[d], Wsk[d],
                                              cvf[d], sf[d]);

    k_aggregate<<<P / 16, 256, 0, stream>>>(x0, x1, x2, points, pidx, mask,
                                            Y0, Y1, Y2, X0, X1, X2);

    // deg0: 1024 tiles / 4 waves = 256 blocks; deg1: 3072/3 = 1024;
    // deg2: 5120/5 = 1024.  All exact -> no tail, EXEC full for WMMA.
    k_gemm_fused<1, 4><<<256,  4 * 32, 0, stream>>>(Y0, X0, cvf[0], sf[0],
                                                    mask, out0);
    k_gemm_fused<3, 3><<<1024, 3 * 32, 0, stream>>>(Y1, X1, cvf[1], sf[1],
                                                    mask, out1);
    k_gemm_fused<5, 5><<<1024, 5 * 32, 0, stream>>>(Y2, X2, cvf[2], sf[2],
                                                    mask, out2);
}
